// CrossAttentionFusion_6923487282332
// MI455X (gfx1250) — matrix operand order repaired and checked by reading, not compiled, not run on hardware
//
#include <hip/hip_runtime.h>

typedef __bf16 v16bf __attribute__((ext_vector_type(16)));
typedef float v8f __attribute__((ext_vector_type(8)));
typedef unsigned int v8u __attribute__((ext_vector_type(8)));

#define Dm 256
#define BT 16      // batch elements per workgroup
#define RT 64      // token rows per workgroup (BT*4)
#define SX 257     // f32 stride of x tile (padded: 257 mod 64 = 1 -> conflict-free)
#define SHW 129    // u32 stride of bf16 A-operand buffers (129 mod 64 = 1)
#define SHH 258    // same in 16-bit units
#define SQW 385    // u32 stride of qkv bf16 buffer (768 bf16 + pad)
#define SQH 770    // same in 16-bit units

static __device__ __forceinline__ unsigned short f2bf(float f) {
  unsigned u = __builtin_bit_cast(unsigned, f);
  u += 0x7FFFu + ((u >> 16) & 1u);   // round-to-nearest-even
  return (unsigned short)(u >> 16);
}
static __device__ __forceinline__ float bf2f(unsigned short h) {
  unsigned u = ((unsigned)h) << 16;
  return __builtin_bit_cast(float, u);
}

// ---------------------------------------------------------------------------
// Weight repack: f32 [K][N] row-major -> bf16 WMMA B-fragment layout.
// Fragment dword index: ((nt*KT + kt)*32 + lane)*8 + v
// B layout used here (16-bit, 32x16, contiguous k): lane n = lane&15, k0 = (lane>>4)*16 + 2v
// ---------------------------------------------------------------------------
__global__ void pack_w(const float* __restrict__ src, unsigned* __restrict__ dst,
                       int K, int N) {
  int fid = blockIdx.x * blockDim.x + threadIdx.x;
  int KT = K >> 5;
  int total = (K * N) >> 1;
  if (fid >= total) return;
  int v = fid & 7;
  int lane = (fid >> 3) & 31;
  int t = fid >> 8;
  int kt = t % KT, nt = t / KT;
  int n = nt * 16 + (lane & 15);
  int k0 = kt * 32 + ((lane >> 4) << 4) + (v << 1);
  unsigned lo = f2bf(src[k0 * N + n]);
  unsigned hi = f2bf(src[(k0 + 1) * N + n]);
  dst[fid] = (hi << 16) | lo;
}

// ---------------------------------------------------------------------------
// Per-wave GEMM stripe: 16 rows x (jend-jbeg)*16 cols, K = KT*32, bf16 WMMA.
// A operand read from LDS buffer (stride SHW dwords) with row = lane&15 and contiguous k = 16*(lane>>4)+s
// (same k order as pack_w); B fragments streamed from L2-resident packed weights.
// ---------------------------------------------------------------------------
template <int KT, typename E>
static __device__ __forceinline__ void wgemm(const unsigned* __restrict__ wf,
                                             const unsigned* abuf, int m0,
                                             int ahalf, int arow, int lane,
                                             int jbeg, int jend, E&& epi) {
  const v8u* wv = (const v8u*)wf;
  for (int c = jbeg; c < jend; c += 8) {
    v8f acc[8] = {};
#pragma unroll
    for (int kt = 0; kt < KT; ++kt) {
      v8u a;
      int base = arow * SHW + kt * 16;
#pragma unroll
      for (int v = 0; v < 8; ++v)
        a[v] = abuf[base + ahalf * 8 + v];
      v16bf av = __builtin_bit_cast(v16bf, a);
#pragma unroll
      for (int j = 0; j < 8; ++j) {
        v8u bw = wv[((c + j) * KT + kt) * 32 + lane];
        v16bf bv = __builtin_bit_cast(v16bf, bw);
        acc[j] = __builtin_amdgcn_wmma_f32_16x16x32_bf16(
            false, av, false, bv, (short)0, acc[j], false, false);
      }
    }
#pragma unroll
    for (int j = 0; j < 8; ++j) {
      int n = (c + j) * 16 + (lane & 15);
#pragma unroll
      for (int v = 0; v < 8; ++v) epi(m0 + (ahalf << 3) + v, n, acc[j][v]);
    }
  }
}

// LayerNorm over all RT rows of the f32 x tile; 4 threads per row.
// TOBF: write bf16 into A-operand buffer; else in-place f32.
template <bool TOBF>
static __device__ void layer_norm_rows(float* xf, unsigned short* dst,
                                       const float* __restrict__ g,
                                       const float* __restrict__ b, float* red,
                                       float* sm, float* sv) {
  const int tid = threadIdx.x;
  const int row = tid >> 2, part = tid & 3;
  float* r = xf + row * SX + part * 64;
  float s = 0.f, q = 0.f;
  for (int cc = 0; cc < 64; ++cc) { float x = r[cc]; s += x; q += x * x; }
  red[row * 8 + part] = s;
  red[row * 8 + 4 + part] = q;
  __syncthreads();
  if (part == 0) {
    float S = red[row * 8] + red[row * 8 + 1] + red[row * 8 + 2] + red[row * 8 + 3];
    float Q = red[row * 8 + 4] + red[row * 8 + 5] + red[row * 8 + 6] + red[row * 8 + 7];
    float m = S * (1.f / 256.f);
    sm[row] = m;
    sv[row] = rsqrtf(Q * (1.f / 256.f) - m * m + 1e-5f);
  }
  __syncthreads();
  float m = sm[row], rs = sv[row];
  const float* gg = g + part * 64;
  const float* bb = b + part * 64;
  if (TOBF) {
    unsigned short* d = dst + row * SHH + part * 64;
    for (int cc = 0; cc < 64; ++cc) d[cc] = f2bf((r[cc] - m) * rs * gg[cc] + bb[cc]);
  } else {
    for (int cc = 0; cc < 64; ++cc) r[cc] = (r[cc] - m) * rs * gg[cc] + bb[cc];
  }
  __syncthreads();
}

// ---------------------------------------------------------------------------
// Fully fused model: embed -> 3 transformer layers -> final LN -> mean -> LN.
// One workgroup (8 wave32) owns 16 batch elements; all activations in LDS.
// ---------------------------------------------------------------------------
__global__ __launch_bounds__(256, 1) void fused_tf(
    const float* __restrict__ global_emb, const float* __restrict__ pert_emb,
    const float* __restrict__ sym_feat, const float* __restrict__ ppi_feat,
    const float* __restrict__ sym_Wm, const float* __restrict__ sym_b,
    const float* __restrict__ sym_ln_g, const float* __restrict__ sym_ln_b,
    const float* __restrict__ tte, const float* __restrict__ bqkv,
    const float* __restrict__ bo, const float* __restrict__ ln1_g,
    const float* __restrict__ ln1_b, const float* __restrict__ ln2_g,
    const float* __restrict__ ln2_b, const float* __restrict__ b1,
    const float* __restrict__ b2, const float* __restrict__ fin_g,
    const float* __restrict__ fin_b, const float* __restrict__ out_g,
    const float* __restrict__ out_b, const unsigned* __restrict__ wsU,
    float* __restrict__ out) {
  __shared__ float xf[RT * SX];       // residual stream, f32
  __shared__ unsigned hb[RT * SHW];   // bf16 A-operand buffer (LN out / attn out)
  __shared__ unsigned qb[RT * SQW];   // bf16 qkv buffer / gelu buffer / scratch
  __shared__ float red[RT * 8];
  __shared__ float sm[RT];
  __shared__ float sv[RT];

  const int tid = threadIdx.x;
  const int lane = tid & 31;
  const int w = tid >> 5;          // wave id, 0..7
  const int m0 = (w >> 1) * 16;    // 16-row M stripe per wave pair
  const int half = w & 1;          // N-half per wave
  const int ahalf = lane >> 4;
  const int arow = m0 + (lane & 15);
  const int b0 = blockIdx.x * BT;

  // ---- embed: sym projection + LN, stack 4 tokens, add token-type emb ----
  float* symt = (float*)qb;  // [16][64] staging
  for (int i = tid; i < BT * 64; i += 256)
    symt[i] = sym_feat[(b0 + (i >> 6)) * 64 + (i & 63)];
  __syncthreads();
  for (int i = tid; i < BT * Dm; i += 256) {
    int bl = i >> 8, cc = i & 255;
    float acc = sym_b[cc];
    const float* s = symt + bl * 64;
#pragma unroll 8
    for (int k = 0; k < 64; ++k) acc += s[k] * sym_Wm[k * Dm + cc];
    xf[(bl * 4 + 2) * SX + cc] = acc;
    xf[(bl * 4 + 0) * SX + cc] = global_emb[(b0 + bl) * Dm + cc] + tte[cc];
    xf[(bl * 4 + 1) * SX + cc] = pert_emb[(b0 + bl) * Dm + cc] + tte[Dm + cc];
    xf[(bl * 4 + 3) * SX + cc] = ppi_feat[(b0 + bl) * Dm + cc] + tte[3 * Dm + cc];
  }
  __syncthreads();
  if (tid < BT) {  // LN over sym rows
    const float* r = xf + (tid * 4 + 2) * SX;
    float s = 0.f, q = 0.f;
    for (int cc = 0; cc < Dm; ++cc) { float x = r[cc]; s += x; q += x * x; }
    float m = s * (1.f / 256.f);
    sm[tid] = m;
    sv[tid] = rsqrtf(q * (1.f / 256.f) - m * m + 1e-5f);
  }
  __syncthreads();
  for (int i = tid; i < BT * Dm; i += 256) {
    int bl = i >> 8, cc = i & 255;
    float* r = &xf[(bl * 4 + 2) * SX + cc];
    *r = (*r - sm[bl]) * sv[bl] * sym_ln_g[cc] + sym_ln_b[cc] + tte[2 * Dm + cc];
  }
  __syncthreads();

  // ---- 3 transformer layers ----
  for (int l = 0; l < 3; ++l) {
    const unsigned* WQ = wsU + l * 98304;
    const unsigned* WOf = wsU + 294912 + l * 32768;
    const unsigned* W1f = wsU + 393216 + l * 32768;
    const unsigned* W2f = wsU + 491520 + l * 32768;
    const float* bq = bqkv + l * 768;
    const float* bol = bo + l * Dm;
    const float* b1l = b1 + l * Dm;
    const float* b2l = b2 + l * Dm;

    // h = LN1(x) -> hb (bf16)
    layer_norm_rows<true>(xf, (unsigned short*)hb, ln1_g + l * Dm, ln1_b + l * Dm,
                          red, sm, sv);

    // qkv = h @ Wqkv + bqkv   (48 N-tiles, split over wave pairs)
    {
      unsigned short* qh = (unsigned short*)qb;
      wgemm<8>(WQ, hb, m0, ahalf, arow, lane, half * 24, half * 24 + 24,
               [&](int m, int n, float val) { qh[m * SQH + n] = f2bf(val + bq[n]); });
    }
    __syncthreads();

    // attention: 16 batch x 8 heads = 128 threads; 4x4 scores per head
    if (tid < 128) {
      int bl = tid >> 3, h = tid & 7;
      const unsigned short* qh = (const unsigned short*)qb;
      unsigned short* oh = (unsigned short*)hb;
      int r0 = bl * 4;
      float sc[4][4];
#pragma unroll
      for (int i = 0; i < 4; ++i) {
        const unsigned short* qq = qh + (r0 + i) * SQH + h * 32;
#pragma unroll
        for (int j = 0; j < 4; ++j) {
          const unsigned short* kk = qh + (r0 + j) * SQH + 256 + h * 32;
          float d = 0.f;
#pragma unroll 8
          for (int c = 0; c < 32; ++c) d += bf2f(qq[c]) * bf2f(kk[c]);
          sc[i][j] = d * 0.17677669529663687f;  // 1/sqrt(32)
        }
      }
#pragma unroll
      for (int i = 0; i < 4; ++i) {
        float mx = fmaxf(fmaxf(sc[i][0], sc[i][1]), fmaxf(sc[i][2], sc[i][3]));
        float e0 = __expf(sc[i][0] - mx), e1 = __expf(sc[i][1] - mx);
        float e2 = __expf(sc[i][2] - mx), e3 = __expf(sc[i][3] - mx);
        float inv = 1.f / (e0 + e1 + e2 + e3);
        sc[i][0] = e0 * inv; sc[i][1] = e1 * inv;
        sc[i][2] = e2 * inv; sc[i][3] = e3 * inv;
      }
#pragma unroll
      for (int i = 0; i < 4; ++i) {
        for (int c = 0; c < 32; ++c) {
          float o = 0.f;
#pragma unroll
          for (int j = 0; j < 4; ++j)
            o += sc[i][j] * bf2f(qh[(r0 + j) * SQH + 512 + h * 32 + c]);
          oh[(r0 + i) * SHH + h * 32 + c] = f2bf(o);
        }
      }
    }
    __syncthreads();

    // x += o @ Wo + bo
    wgemm<8>(WOf, hb, m0, ahalf, arow, lane, half * 8, half * 8 + 8,
             [&](int m, int n, float val) { xf[m * SX + n] += val + bol[n]; });
    __syncthreads();

    // h2 = LN2(x) -> hb
    layer_norm_rows<true>(xf, (unsigned short*)hb, ln2_g + l * Dm, ln2_b + l * Dm,
                          red, sm, sv);

    // g = gelu(h2 @ W1 + b1) -> qb (as A-operand layout)
    {
      unsigned short* gh = (unsigned short*)qb;
      wgemm<8>(W1f, hb, m0, ahalf, arow, lane, half * 8, half * 8 + 8,
               [&](int m, int n, float val) {
                 float x = val + b1l[n];
                 gh[m * SHH + n] = f2bf(0.5f * x * (1.f + erff(x * 0.70710678118654752f)));
               });
    }
    __syncthreads();

    // x += g @ W2 + b2
    wgemm<8>(W2f, qb, m0, ahalf, arow, lane, half * 8, half * 8 + 8,
             [&](int m, int n, float val) { xf[m * SX + n] += val + b2l[n]; });
    __syncthreads();
  }

  // ---- final LN (in place), token mean, out LN, store ----
  layer_norm_rows<false>(xf, nullptr, fin_g, fin_b, red, sm, sv);
  float* ob = (float*)qb;  // [16][256] mean rows
  if (tid < BT) {
    int bl = tid;
    float s = 0.f, q = 0.f;
    for (int cc = 0; cc < Dm; ++cc) {
      float mval = 0.25f * (xf[(bl * 4 + 0) * SX + cc] + xf[(bl * 4 + 1) * SX + cc] +
                            xf[(bl * 4 + 2) * SX + cc] + xf[(bl * 4 + 3) * SX + cc]);
      ob[bl * Dm + cc] = mval;
      s += mval;
      q += mval * mval;
    }
    float m = s * (1.f / 256.f);
    sm[bl] = m;
    sv[bl] = rsqrtf(q * (1.f / 256.f) - m * m + 1e-5f);
  }
  __syncthreads();
  for (int i = tid; i < BT * Dm; i += 256) {
    int bl = i >> 8, cc = i & 255;
    out[(b0 + bl) * Dm + cc] =
        (ob[bl * Dm + cc] - sm[bl]) * sv[bl] * out_g[cc] + out_b[cc];
  }
}

extern "C" void kernel_launch(void* const* d_in, const int* in_sizes, int n_in,
                              void* d_out, int out_size, void* d_ws, size_t ws_size,
                              hipStream_t stream) {
  (void)n_in; (void)out_size; (void)ws_size;
  const float* global_emb = (const float*)d_in[0];
  const float* pert_emb = (const float*)d_in[1];
  const float* sym_feat = (const float*)d_in[2];
  const float* ppi_feat = (const float*)d_in[3];
  const float* sym_W = (const float*)d_in[4];
  const float* sym_b = (const float*)d_in[5];
  const float* sym_ln_g = (const float*)d_in[6];
  const float* sym_ln_b = (const float*)d_in[7];
  const float* tte = (const float*)d_in[8];
  const float* Wqkv = (const float*)d_in[9];
  const float* bqkv = (const float*)d_in[10];
  const float* Wo = (const float*)d_in[11];
  const float* bo = (const float*)d_in[12];
  const float* ln1_g = (const float*)d_in[13];
  const float* ln1_b = (const float*)d_in[14];
  const float* ln2_g = (const float*)d_in[15];
  const float* ln2_b = (const float*)d_in[16];
  const float* W1 = (const float*)d_in[17];
  const float* b1 = (const float*)d_in[18];
  const float* W2 = (const float*)d_in[19];
  const float* b2 = (const float*)d_in[20];
  const float* fin_g = (const float*)d_in[21];
  const float* fin_b = (const float*)d_in[22];
  const float* out_g = (const float*)d_in[23];
  const float* out_b = (const float*)d_in[24];
  float* out = (float*)d_out;
  unsigned* wsU = (unsigned*)d_ws;

  // Repack weights into bf16 WMMA-fragment layout (L2-resident, ~2.25 MB).
  for (int l = 0; l < 3; ++l) {
    pack_w<<<384, 256, 0, stream>>>(Wqkv + l * 196608, wsU + l * 98304, 256, 768);
    pack_w<<<128, 256, 0, stream>>>(Wo + l * 65536, wsU + 294912 + l * 32768, 256, 256);
    pack_w<<<128, 256, 0, stream>>>(W1 + l * 65536, wsU + 393216 + l * 32768, 256, 256);
    pack_w<<<128, 256, 0, stream>>>(W2 + l * 65536, wsU + 491520 + l * 32768, 256, 256);
  }

  int Bsz = in_sizes[0] / Dm;  // 32768
  fused_tf<<<Bsz / BT, 256, 0, stream>>>(
      global_emb, pert_emb, sym_feat, ppi_feat, sym_W, sym_b, sym_ln_g, sym_ln_b,
      tte, bqkv, bo, ln1_g, ln1_b, ln2_g, ln2_b, b1, b2, fin_g, fin_b, out_g,
      out_b, wsU, out);
}
